// multi_MLP_36292473651990
// MI455X (gfx1250) — matrix operand order repaired and checked by reading, not compiled, not run on hardware
//
#include <hip/hip_runtime.h>
#include <hip/hip_bf16.h>

// ---------------- problem constants (from reference) ----------------
#define N_PTS   262144
#define N_HEADS 16
#define HID     256
#define N_LAY   3          // hidden layers
#define CAP     (N_PTS / N_HEADS)   // 16384
#define PE_L    10
#define PE_DIM  40
#define K0      64         // PE_DIM padded to multiple of 32
#define TPTS    128        // points per workgroup (M tile)
#define NOUT    16         // OUT_F=3 padded to one WMMA N-tile

typedef __attribute__((ext_vector_type(16))) __bf16 v16bf;
typedef __attribute__((ext_vector_type(8)))  __bf16 v8bf;
typedef __attribute__((ext_vector_type(8)))  float  v8f;

// ---------------- pass 1: f32 -> bf16 weight conversion ----------------
// win_bf : [H][64][256]   (rows 40..63 zero-padded)
// whid_bf: [H][3][256][256]
// wout_bf: [H][256][16]   (cols 3..15 zero-padded)
__global__ __launch_bounds__(256) void cvt_weights_kernel(
    const float* __restrict__ Win, const float* __restrict__ Whid,
    const float* __restrict__ Wout,
    __bf16* __restrict__ win_bf, __bf16* __restrict__ whid_bf,
    __bf16* __restrict__ wout_bf)
{
    const int i = blockIdx.x * 256 + threadIdx.x;
    if (i < N_HEADS * K0 * HID) {
        const int n = i & (HID - 1);
        const int k = (i >> 8) & (K0 - 1);
        const int h = i >> 14;
        float v = (k < PE_DIM) ? Win[(h * PE_DIM + k) * HID + n] : 0.0f;
        win_bf[i] = (__bf16)v;
    }
    if (i < N_HEADS * N_LAY * HID * HID) {
        whid_bf[i] = (__bf16)Whid[i];
    }
    if (i < N_HEADS * HID * NOUT) {
        const int nn = i & (NOUT - 1);
        const int k  = (i >> 4) & (HID - 1);
        const int h  = i >> 12;
        float v = (nn < 3) ? Wout[(h * HID + k) * 3 + nn] : 0.0f;
        wout_bf[i] = (__bf16)v;
    }
}

// ---------------- async global -> LDS copy (ASYNCcnt-tracked) ----------------
__device__ __forceinline__ void async_copy_to_lds(const void* gsrc, void* ldst,
                                                  int bytes, int tid)
{
    const char* g = (const char*)gsrc;
    char*       d = (char*)ldst;
    for (int i = tid * 16; i < bytes; i += 256 * 16) {
        uint64_t ga = (uint64_t)(uintptr_t)(g + i);
        uint32_t la = (uint32_t)(uintptr_t)(d + i);   // low 32 bits = LDS address
        asm volatile("global_load_async_to_lds_b128 %0, %1, off"
                     :: "v"(la), "v"(ga) : "memory");
    }
}

// ---------------- WMMA layer: act[128x256] (bf16, LDS) x wbuf[KT*32 x 256] ----------------
// Each wave owns rows m0..m0+15.  A-fragments are loaded to registers first,
// so the wave can safely overwrite its own rows of `act` with the layer output.
template <int KT>
__device__ __forceinline__ void mlp_layer(__bf16* act, const __bf16* wbuf,
                                          const float* __restrict__ bias,
                                          int wave, int lane)
{
    const int m0 = wave * 16;
    const int nl = lane & 15;        // A: row within tile / C: column / B: column
    const int hf = lane >> 4;        // half selector

    // ---- load A fragments (16-bit A layout: lane = M, VGPR pairs = K) ----
    v16bf a[KT];
#pragma unroll
    for (int kt = 0; kt < KT; ++kt) {
        const __bf16* ar = act + (m0 + nl) * HID + kt * 32 + hf * 8;
        v8bf lo = *(const v8bf*)(ar);        // K = base+0..7   -> frag[0..7]
        v8bf hi = *(const v8bf*)(ar + 16);   // K = base+16..23 -> frag[8..15]
#pragma unroll
        for (int e = 0; e < 8; ++e) { a[kt][e] = lo[e]; a[kt][e + 8] = hi[e]; }
    }

    // ---- N sweep ----
    for (int nt = 0; nt < 16; ++nt) {
        v8f acc = {};
#pragma unroll
        for (int kt = 0; kt < KT; ++kt) {
            const __bf16* bc = wbuf + (kt * 32 + hf * 8) * HID + nt * 16 + nl;  // B: lane = N
            v16bf b;
#pragma unroll
            for (int e = 0; e < 8; ++e) { b[e] = bc[e * HID]; b[e + 8] = bc[(e + 16) * HID]; }
            acc = __builtin_amdgcn_wmma_f32_16x16x32_bf16(
                false, a[kt], false, b, (short)0, acc, false, false);
        }
        const float bn = bias[nt * 16 + nl];
#pragma unroll
        for (int r = 0; r < 8; ++r) {                  // C layout: lane = N, vgpr = M
            float v = fmaxf(acc[r] + bn, 0.0f);        // bias + ReLU
            act[(m0 + r + hf * 8) * HID + nt * 16 + nl] = (__bf16)v;
        }
    }
}

// ---------------- pass 2: fused gather + PE + MLP + scatter ----------------
__global__ __launch_bounds__(256) void multi_mlp_kernel(
    const float* __restrict__ coords,      // [N_PTS][2]
    const int*   __restrict__ head_idx,    // [H][CAP]
    const __bf16* __restrict__ win_bf,     // [H][64][256]
    const __bf16* __restrict__ whid_bf,    // [H][3][256][256]
    const __bf16* __restrict__ wout_bf,    // [H][256][16]
    const float* __restrict__ b_in,        // [H][256]
    const float* __restrict__ b_hid,       // [H][3][256]
    const float* __restrict__ b_out,       // [H][3]
    float*       __restrict__ out)         // [N_PTS][3]
{
    extern __shared__ char smem_raw[];
    __bf16* buf0 = (__bf16*)smem_raw;                                // 128 KB
    __bf16* buf1 = (__bf16*)(smem_raw + HID * HID * 2);              // 128 KB
    __bf16* act  = (__bf16*)(smem_raw + 2 * HID * HID * 2);          //  64 KB
    __bf16* wb[2] = { buf0, buf1 };

    const int tile = blockIdx.x;           // 0..127
    const int h    = blockIdx.y;           // 0..15
    const int tid  = threadIdx.x;
    const int lane = tid & 31;
    const int wave = tid >> 5;

    const __bf16* whid_h = whid_bf + (size_t)h * N_LAY * HID * HID;

    // ---- prologue: async-prefetch W_in -> buf0 (8 chunks/thr) and
    //      W_hid[0] -> buf1 (32 chunks/thr); they complete in order ----
    async_copy_to_lds(win_bf + (size_t)h * K0 * HID, buf0, K0 * HID * 2, tid);
    async_copy_to_lds(whid_h, buf1, HID * HID * 2, tid);

    // ---- gather + positional encoding into act[p][0..63] (overlaps copies) ----
    if (tid < TPTS) {
        const int gid = head_idx[h * CAP + tile * TPTS + tid];
        const float x = coords[gid * 2 + 0];
        const float y = coords[gid * 2 + 1];
        __bf16* row = act + tid * HID;
        float f = 3.14159265358979323846f;
#pragma unroll
        for (int l = 0; l < PE_L; ++l) {
            const float ax = x * f, ay = y * f;
            row[4 * l + 0] = (__bf16)sinf(ax);
            row[4 * l + 1] = (__bf16)sinf(ay);
            row[4 * l + 2] = (__bf16)cosf(ax);
            row[4 * l + 3] = (__bf16)cosf(ay);
            f *= 2.0f;
        }
        for (int k = PE_DIM; k < K0; ++k) row[k] = (__bf16)0.0f;
    }

    // wait only for the W_in chunks (first 8 of this thread's 40 async ops)
    asm volatile("s_wait_asynccnt 32" ::: "memory");
    __syncthreads();

    mlp_layer<K0 / 32>(act, buf0, b_in + h * HID, wave, lane);

    // ---- hidden layers with double-buffered async weight prefetch ----
    for (int l = 0; l < N_LAY; ++l) {
        __syncthreads();                                  // buffer wb[l&1] free
        asm volatile("s_wait_asynccnt 0" ::: "memory");   // layer-l panel arrived
        if (l < N_LAY - 1)
            async_copy_to_lds(whid_h + (size_t)(l + 1) * HID * HID,
                              wb[l & 1], HID * HID * 2, tid);
        __syncthreads();                                  // arrival visible to all
        mlp_layer<HID / 32>(act, wb[(l + 1) & 1],
                            b_hid + (h * N_LAY + l) * HID, wave, lane);
    }

    // ---- output layer (256 -> 3, padded to 16) via WMMA; no barrier needed:
    //      each wave reads only its own 16 rows of act ----
    {
        const int m0 = wave * 16;
        const int nl = lane & 15;
        const int hf = lane >> 4;
        const __bf16* wo = wout_bf + (size_t)h * HID * NOUT;

        v8f acc = {};
#pragma unroll
        for (int kt = 0; kt < HID / 32; ++kt) {
            const __bf16* ar = act + (m0 + nl) * HID + kt * 32 + hf * 8;
            v8bf lo = *(const v8bf*)(ar);
            v8bf hi = *(const v8bf*)(ar + 16);
            v16bf a;
#pragma unroll
            for (int e = 0; e < 8; ++e) { a[e] = lo[e]; a[e + 8] = hi[e]; }
            const __bf16* bc = wo + (kt * 32 + hf * 8) * NOUT + nl;
            v16bf b;
#pragma unroll
            for (int e = 0; e < 8; ++e) { b[e] = bc[e * NOUT]; b[e + 8] = bc[(e + 16) * NOUT]; }
            acc = __builtin_amdgcn_wmma_f32_16x16x32_bf16(
                false, a, false, b, (short)0, acc, false, false);
        }
        if (nl < 3) {
            const float bo = b_out[h * 3 + nl];
            const int* hix = head_idx + h * CAP + tile * TPTS + m0;
#pragma unroll
            for (int r = 0; r < 8; ++r) {                // lane = N, vgpr = M
                const int gid = hix[r + 8 * hf];
                out[(long)gid * 3 + nl] = acc[r] + bo;
            }
        }
    }
}

// ---------------- host launcher ----------------
extern "C" void kernel_launch(void* const* d_in, const int* in_sizes, int n_in,
                              void* d_out, int out_size, void* d_ws, size_t ws_size,
                              hipStream_t stream)
{
    (void)in_sizes; (void)n_in; (void)out_size; (void)ws_size;

    const float* coords   = (const float*)d_in[0];
    const int*   head_idx = (const int*)d_in[1];
    const float* W_in     = (const float*)d_in[2];
    const float* b_in     = (const float*)d_in[3];
    const float* W_hid    = (const float*)d_in[4];
    const float* b_hid    = (const float*)d_in[5];
    const float* W_out    = (const float*)d_in[6];
    const float* b_out    = (const float*)d_in[7];
    float* out = (float*)d_out;

    // workspace: bf16 weights
    char* ws = (char*)d_ws;
    __bf16* win_bf  = (__bf16*)ws;                                  // 512 KB
    __bf16* whid_bf = (__bf16*)(ws + (size_t)N_HEADS * K0 * HID * 2);         // 6 MB
    __bf16* wout_bf = (__bf16*)(ws + (size_t)N_HEADS * K0 * HID * 2
                                   + (size_t)N_HEADS * N_LAY * HID * HID * 2); // 128 KB

    // pass 1: weight conversion
    {
        const int total = N_HEADS * N_LAY * HID * HID;   // dominant count
        cvt_weights_kernel<<<(total + 255) / 256, 256, 0, stream>>>(
            W_in, W_hid, W_out, win_bf, whid_bf, wout_bf);
    }

    // pass 2: fused MLP.  LDS: 2x128KB weight panels + 64KB act = 320 KB (WGP max)
    {
        dim3 grid(CAP / TPTS, N_HEADS);  // 128 x 16
        const size_t smem = 2 * (size_t)HID * HID * 2 + (size_t)TPTS * HID * 2;
        multi_mlp_kernel<<<grid, 256, smem, stream>>>(
            coords, head_idx, win_bf, whid_bf, wout_bf,
            b_in, b_hid, b_out, out);
    }

    // reference returns (out_final, coords): append coords after out_final
    hipMemcpyAsync(out + (size_t)N_PTS * 3, d_in[0],
                   (size_t)N_PTS * 2 * sizeof(float),
                   hipMemcpyDeviceToDevice, stream);
}
